// DWConv2_45973329936716
// MI455X (gfx1250) — compile-verified, matrix operand order checked by reading, not run on hardware
//
#include <hip/hip_runtime.h>
#include <hip/hip_bf16.h>
#include <math.h>

typedef __attribute__((ext_vector_type(16))) _Float16 v16h;
typedef __attribute__((ext_vector_type(8)))  _Float16 v8h;
typedef __attribute__((ext_vector_type(8)))  float    v8f;

#define HH  128
#define WW  128
#define NB  8
#define CI  64
#define CO  64
#define HW  (HH*WW)

// Workspace element counts
#define OM_ELEMS    ((size_t)NB * 32 * HW)     // fp32 combined: 0-17 offsets,
                                               // 18-26 sigmoid(mask), 27-31 pad
#define BP_ELEMS    (9 * 2 * 4 * 512)          // 36864 f16: deform weight tiles
#define WP2_ELEMS   (9 * 2 * 2 * 512)          // 18432 f16: offset/mask weight tiles
#define XH_ELEMS    ((size_t)NB * HW * 64)     // f16 NHWC x
#define ZB_ELEMS    64                         // zeroed f16 pad row

// ---------------------------------------------------------------------------
// Kernel 1: pack weights into f16 WMMA B-tiles (+ zero the OOB pad row).
// B-tile lane layout mirrors 16-bit A layout: lane L<16 -> N=L, K in {0-7,16-23};
// lane L>=16 -> N=L-16, K in {8-15,24-31}; element e -> K = khalf + (e<8?e:e+8).
//  - Bp : deform conv_w, tiles (k,q,t) t=0..3   -> N = couts t*16..
//  - Wp2: offset_w(18) ++ mask_w(9) ++ zeros,   tiles (k,q,t2) t2=0..1
// ---------------------------------------------------------------------------
__global__ void pack_w_kernel(const float* __restrict__ conv_w,
                              const float* __restrict__ offset_w,
                              const float* __restrict__ mask_w,
                              _Float16* __restrict__ Bp,
                              _Float16* __restrict__ Wp2,
                              _Float16* __restrict__ zbuf) {
  int idx  = blockIdx.x * 256 + threadIdx.x;     // 0..55295
  if (idx < ZB_ELEMS) zbuf[idx] = (_Float16)0.0f;
  int e    = idx & 15;
  int lane = (idx >> 4) & 31;
  int nrow = lane & 15;
  int khalf = (lane >> 4) ? 8 : 0;
  if (idx < BP_ELEMS) {
    int ti = idx >> 9;                           // (k*2+q)*4 + t
    int t  = ti & 3;
    int q  = (ti >> 2) & 1;
    int k  = ti >> 3;
    int o  = t * 16 + nrow;
    int c  = q * 32 + khalf + (e < 8 ? e : e + 8);
    Bp[idx] = (_Float16)conv_w[(o * 64 + c) * 9 + k];
  } else {
    int i2 = idx - BP_ELEMS;                     // 0..18431
    int ti = i2 >> 9;                            // (k*2+q)*2 + t2
    int t2 = ti & 1;
    int q  = (ti >> 1) & 1;
    int k  = ti >> 2;
    int j  = t2 * 16 + nrow;                     // output channel (27 used)
    int c  = q * 32 + khalf + (e < 8 ? e : e + 8);
    float v = 0.0f;
    if (j < 18)      v = offset_w[(j * 64 + c) * 9 + k];
    else if (j < 27) v = mask_w[((j - 18) * 64 + c) * 9 + k];
    Wp2[i2] = (_Float16)v;
  }
}

// ---------------------------------------------------------------------------
// Kernel 2: x NCHW fp32 -> NHWC f16 via LDS tile transpose (64 pix x 64 ch).
// ---------------------------------------------------------------------------
__global__ void nhwc_kernel(const float* __restrict__ x, _Float16* __restrict__ xh) {
  __shared__ _Float16 s[64][72];                 // padded to dodge bank conflicts
  int tid  = threadIdx.x;
  int blk  = blockIdx.x;                         // 2048 blocks
  int n    = blk >> 8;                           // HW/64 = 256 blocks per image
  int pix0 = (blk & 255) * 64;
  const float* xb = x + (size_t)n * CI * HW + pix0;
  for (int i = tid; i < 4096; i += 256) {        // coalesced along pixel dim
    int c  = i >> 6;
    int pp = i & 63;
    s[pp][c] = (_Float16)xb[c * HW + pp];
  }
  __syncthreads();
  _Float16* ob = xh + ((size_t)n * HW + pix0) * 64;
  for (int i = tid; i < 4096; i += 256) {        // coalesced along channel dim
    int pp = i >> 6;
    int c  = i & 63;
    ob[pp * 64 + c] = s[pp][c];
  }
}

// ---------------------------------------------------------------------------
// Kernel 3: offset(18) + mask(9,sigmoid) 3x3 convs as WMMA GEMM.
// Wave = 16 pixels x 32 outputs (27 used). K = 9 taps x 64 ch. Per tap:
// base-pointer select against a zeroed pad row, 2 K-chunks x
// (2 b128 loads + 1 WMMA per N-tile). Branch-free epilogue: identity-vs-
// sigmoid chosen by cndmask, all 32 channels stored unconditionally.
// ---------------------------------------------------------------------------
__global__ void __launch_bounds__(256)
offmask_wmma_kernel(const _Float16* __restrict__ xh,
                    const float* __restrict__ ob, const float* __restrict__ mb,
                    const _Float16* __restrict__ Wp2,
                    const _Float16* __restrict__ zbuf,
                    float* __restrict__ om) {
  __shared__ _Float16 sW[WP2_ELEMS];             // 36 KB
  for (int i = threadIdx.x; i < WP2_ELEMS / 8; i += 256)
    ((v8h*)sW)[i] = ((const v8h*)Wp2)[i];
  __syncthreads();

  int wave  = (int)((blockIdx.x * 256 + threadIdx.x) >> 5);  // 0..8191
  int lane  = threadIdx.x & 31;
  int p     = lane & 15;
  int khalf = (lane >> 4) * 8;
  int w0    = (wave & 7) * 16;
  int hq    = (wave >> 3) & 127;
  int n     = wave >> 10;
  int wp    = w0 + p;

  const _Float16* xn = xh + (size_t)n * HW * 64;

  v8f acc[2];
#pragma unroll
  for (int t2 = 0; t2 < 2; ++t2) {
    int j = t2 * 16 + p;
    float b = (j < 18) ? ob[j] : ((j < 27) ? mb[j - 18] : 0.0f);
#pragma unroll
    for (int r = 0; r < 8; ++r) acc[t2][r] = b;
  }

#pragma unroll 1
  for (int k = 0; k < 9; ++k) {
    int kh = k / 3, kw = k % 3;
    int yy = hq - 1 + kh, xx = wp - 1 + kw;
    bool ok = (yy >= 0 && yy < HH && xx >= 0 && xx < WW);
    const _Float16* src = ok ? (xn + (yy * WW + xx) * 64) : zbuf;  // ptr select
#pragma unroll
    for (int q = 0; q < 2; ++q) {
      int c0 = q * 32 + khalf;
      v8h ra = *(const v8h*)(src + c0);
      v8h rb = *(const v8h*)(src + c0 + 16);
      v16h a = __builtin_shufflevector(ra, rb, 0, 1, 2, 3, 4, 5, 6, 7,
                                       8, 9, 10, 11, 12, 13, 14, 15);
      const _Float16* wpp = sW + ((k * 2 + q) * 2) * 512 + lane * 16;
#pragma unroll
      for (int t2 = 0; t2 < 2; ++t2) {
        v16h bt = *(const v16h*)(wpp + t2 * 512);
        acc[t2] = __builtin_amdgcn_wmma_f32_16x16x32_f16(
            false, a, false, bt, (short)0, acc[t2], false, false);
      }
    }
  }

  // Branch-free store: channels 0-17 raw, 18-26 sigmoid, 27-31 pad garbage.
  float* po = om + (size_t)n * 32 * HW + hq * WW + w0;
#pragma unroll
  for (int t2 = 0; t2 < 2; ++t2) {
    int j = t2 * 16 + p;
    bool is_mask = (j >= 18);
#pragma unroll
    for (int r = 0; r < 8; ++r) {
      int col = r + khalf;                       // pixel M = r + khalf
      float v = acc[t2][r];
      float s = 1.0f / (1.0f + expf(-v));
      po[j * HW + col] = is_mask ? s : v;
    }
  }
}

// ---------------------------------------------------------------------------
// Kernel 4: modulated deformable conv via WMMA, NHWC f16 gathers.
// Wave = 16 pixels x 64 couts. Per tap: 4 bilinear corners, each 2x16B
// contiguous loads per lane; 4-corner lerp done in packed f16 (pk_fma);
// B-tiles served from LDS.
// ---------------------------------------------------------------------------
__global__ void __launch_bounds__(256)
deform_wmma_kernel(const _Float16* __restrict__ xh,
                   const float* __restrict__ om,
                   const _Float16* __restrict__ Bp,
                   const float* __restrict__ conv_b,
                   float* __restrict__ out) {
  extern __shared__ _Float16 sB[];               // 72 KB (dynamic, >64KB opt-in)
  for (int i = threadIdx.x; i < BP_ELEMS / 8; i += 256)
    ((v8h*)sB)[i] = ((const v8h*)Bp)[i];
  __syncthreads();

  int wave  = (int)((blockIdx.x * 256 + threadIdx.x) >> 5);  // 0..8191
  int lane  = threadIdx.x & 31;
  int p     = lane & 15;
  int khalf = (lane >> 4) * 8;
  int w0    = (wave & 7) * 16;
  int hq    = (wave >> 3) & 127;
  int n     = wave >> 10;
  int wp    = w0 + p;

  const _Float16* xn = xh + (size_t)n * HW * 64;
  const float* offb  = om + (size_t)n * 32 * HW + hq * WW + wp;

  v8f acc[4];
#pragma unroll
  for (int t = 0; t < 4; ++t) {
    float b = conv_b[t * 16 + p];
#pragma unroll
    for (int r = 0; r < 8; ++r) acc[t][r] = b;
  }

#pragma unroll 1
  for (int k = 0; k < 9; ++k) {
    int kh = k / 3, kw = k % 3;
    float offy = offb[(2 * k) * HW];
    float offx = offb[(2 * k + 1) * HW];
    float m    = offb[(18 + k) * HW];

    float py = (float)(hq - 1 + kh) + offy;
    float px = (float)(wp - 1 + kw) + offx;
    float y0f = floorf(py), x0f = floorf(px);
    float wy = py - y0f,   wx = px - x0f;
    int y0 = (int)y0f, x0 = (int)x0f;
    int y1 = y0 + 1,   x1 = x0 + 1;
    float vy0 = (y0 >= 0 && y0 < HH) ? 1.0f : 0.0f;
    float vy1 = (y1 >= 0 && y1 < HH) ? 1.0f : 0.0f;
    float vx0 = (x0 >= 0 && x0 < WW) ? 1.0f : 0.0f;
    float vx1 = (x1 >= 0 && x1 < WW) ? 1.0f : 0.0f;
    int yc0 = min(max(y0, 0), HH - 1), yc1 = min(max(y1, 0), HH - 1);
    int xc0 = min(max(x0, 0), WW - 1), xc1 = min(max(x1, 0), WW - 1);
    // fold mask + validity into the four corner weights, then go packed f16
    _Float16 h00 = (_Float16)((1.0f - wy) * (1.0f - wx) * m * vy0 * vx0);
    _Float16 h01 = (_Float16)((1.0f - wy) * wx          * m * vy0 * vx1);
    _Float16 h10 = (_Float16)(wy          * (1.0f - wx) * m * vy1 * vx0);
    _Float16 h11 = (_Float16)(wy          * wx          * m * vy1 * vx1);
    int b00 = (yc0 * WW + xc0) * 64, b01 = (yc0 * WW + xc1) * 64;
    int b10 = (yc1 * WW + xc0) * 64, b11 = (yc1 * WW + xc1) * 64;

#pragma unroll
    for (int q = 0; q < 2; ++q) {
      int c0 = q * 32 + khalf;
      v8h r00a = *(const v8h*)(xn + b00 + c0);
      v8h r00b = *(const v8h*)(xn + b00 + c0 + 16);
      v8h r01a = *(const v8h*)(xn + b01 + c0);
      v8h r01b = *(const v8h*)(xn + b01 + c0 + 16);
      v8h r10a = *(const v8h*)(xn + b10 + c0);
      v8h r10b = *(const v8h*)(xn + b10 + c0 + 16);
      v8h r11a = *(const v8h*)(xn + b11 + c0);
      v8h r11b = *(const v8h*)(xn + b11 + c0 + 16);
      v16h a;
#pragma unroll
      for (int e = 0; e < 8; ++e) {              // vectorizes to v_pk_fma_f16
        a[e]     = (_Float16)(r00a[e] * h00 + r01a[e] * h01
                            + r10a[e] * h10 + r11a[e] * h11);
        a[8 + e] = (_Float16)(r00b[e] * h00 + r01b[e] * h01
                            + r10b[e] * h10 + r11b[e] * h11);
      }
      const _Float16* bp = sB + ((k * 2 + q) * 4) * 512 + lane * 16;
#pragma unroll
      for (int t = 0; t < 4; ++t) {
        v16h bt = *(const v16h*)(bp + t * 512);
        acc[t] = __builtin_amdgcn_wmma_f32_16x16x32_f16(
            false, a, false, bt, (short)0, acc[t], false, false);
      }
    }
  }

  // D layout: VGPR r -> M = r + khalf (pixel), N = p (cout within tile).
  float* ob2 = out + (size_t)(n * CO) * HW + hq * WW + w0;
#pragma unroll
  for (int t = 0; t < 4; ++t) {
    int o = t * 16 + p;
#pragma unroll
    for (int r = 0; r < 8; ++r) {
      ob2[o * HW + (r + khalf)] = acc[t][r];
    }
  }
}

// ---------------------------------------------------------------------------
extern "C" void kernel_launch(void* const* d_in, const int* in_sizes, int n_in,
                              void* d_out, int out_size, void* d_ws, size_t ws_size,
                              hipStream_t stream) {
  const float* x        = (const float*)d_in[0];
  const float* offset_w = (const float*)d_in[1];
  const float* offset_b = (const float*)d_in[2];
  const float* mask_w   = (const float*)d_in[3];
  const float* mask_b   = (const float*)d_in[4];
  const float* conv_w   = (const float*)d_in[5];
  const float* conv_b   = (const float*)d_in[6];
  float* out = (float*)d_out;

  // workspace: om f32 (combined offset+mask) | Bp f16 | Wp2 f16 | xh f16 | zbuf
  float*    omB  = (float*)d_ws;
  _Float16* Bp   = (_Float16*)(omB + OM_ELEMS);
  _Float16* Wp2  = Bp + BP_ELEMS;
  _Float16* xh   = Wp2 + WP2_ELEMS;
  _Float16* zbuf = xh + XH_ELEMS;

  (void)hipFuncSetAttribute((const void*)deform_wmma_kernel,
                            hipFuncAttributeMaxDynamicSharedMemorySize,
                            (int)(BP_ELEMS * sizeof(_Float16)));

  pack_w_kernel<<<216, 256, 0, stream>>>(conv_w, offset_w, mask_w, Bp, Wp2, zbuf);
  nhwc_kernel<<<2048, 256, 0, stream>>>(x, xh);
  offmask_wmma_kernel<<<1024, 256, 0, stream>>>(xh, offset_b, mask_b, Wp2, zbuf,
                                                omB);
  deform_wmma_kernel<<<1024, 256, BP_ELEMS * sizeof(_Float16), stream>>>(
      xh, omB, Bp, conv_b, out);
}
